// PlasticMambaBlock_77386720740105
// MI455X (gfx1250) — matrix operand order repaired and checked by reading, not compiled, not run on hardware
//
#include <hip/hip_runtime.h>

// ---------------- problem constants (match reference) ----------------
#define D_MODEL 1024
#define SEQ_L   2048
#define BATCH   2
#define DSTATE  16
#define DCONV   4
#define DINNER  2048                // 2*D
#define DTRANK  64                  // D/16
#define XDBL_W  (DTRANK + 2*DSTATE) // 96 (logical)
#define XDBL_P  128                 // padded row stride for x_dbl / x_proj_w
#define ETA_C   0.1f
#define LN_DECAY (-0.051293294f)    // ln(0.95)
#define EPS_C   1e-5f

typedef __bf16 bf16;
typedef bf16  v16bf __attribute__((ext_vector_type(16)));
typedef float v8f   __attribute__((ext_vector_type(8)));

__device__ __forceinline__ float siluf(float x) { return x / (1.0f + __expf(-x)); }

// pack 8 fp32 -> 8 bf16 -> one 16B LDS store
__device__ __forceinline__ void pack8_store(float4 a, float4 b, bf16* dst) {
  union { bf16 h[8]; uint4 u; } p;
  p.h[0] = (bf16)a.x; p.h[1] = (bf16)a.y; p.h[2] = (bf16)a.z; p.h[3] = (bf16)a.w;
  p.h[4] = (bf16)b.x; p.h[5] = (bf16)b.y; p.h[6] = (bf16)b.z; p.h[7] = (bf16)b.w;
  *reinterpret_cast<uint4*>(dst) = p.u;
}

// =====================================================================
// RMSNorm: one block per row of length `dim`
// =====================================================================
__global__ __launch_bounds__(256)
void k_rmsnorm(const float* __restrict__ x, const float* __restrict__ w,
               float* __restrict__ o, int rows, int dim) {
  __shared__ float red[8];
  int row = blockIdx.x;
  if (row >= rows) return;
  const float* xr = x + (size_t)row * dim;
  float ss = 0.0f;
  for (int i = threadIdx.x; i < dim; i += 256) { float v = xr[i]; ss += v * v; }
  for (int off = 16; off > 0; off >>= 1) ss += __shfl_down(ss, off, 32);
  if ((threadIdx.x & 31) == 0) red[threadIdx.x >> 5] = ss;
  __syncthreads();
  if (threadIdx.x == 0) {
    float v = 0.0f;
    #pragma unroll
    for (int i = 0; i < 8; ++i) v += red[i];
    red[0] = v;
  }
  __syncthreads();
  float scale = rsqrtf(red[0] / (float)dim + EPS_C);
  float* orow = o + (size_t)row * dim;
  for (int i = threadIdx.x; i < dim; i += 256) orow[i] = xr[i] * scale * w[i];
}

// =====================================================================
// WMMA GEMM (guard-free): C[M,N] = A[M,K] * op(B)
//   REQUIRES M%128==0, N%64==0, K%32==0 (all launches comply; x_proj is
//   padded to N=128).
//   BT=true : B stored [N,K] row-major (NT);  BT=false: B stored [K,N] (NN)
// Block tile 128x64, K-step 32, double-buffered LDS; 8 waves in 4x2,
// wave tile 32x32 -> 4 v_wmma_f32_16x16x32_bf16 per k-step per wave.
// =====================================================================
template <bool BT>
__global__ __launch_bounds__(256)
void k_gemm(const float* __restrict__ A, const float* __restrict__ B,
            float* __restrict__ C, int M, int N, int K,
            int lda, int ldb, int ldc) {
  constexpr int BM = 128, BN = 64, BK = 32, LDSW = 40; // 80B row stride
  __shared__ __align__(16) bf16 sA[2][BM][LDSW];
  __shared__ __align__(16) bf16 sB[2][BN][LDSW];

  const int tid  = threadIdx.x;
  const int lane = tid & 31;
  const int wid  = tid >> 5;
  const int m0 = blockIdx.y * BM;
  const int n0 = blockIdx.x * BN;
  const int wm = (wid >> 1) * 32;   // 0,32,64,96
  const int wn = (wid & 1) * 32;    // 0,32
  const int kg = lane >> 4;
  const int lr = lane & 15;

  // staging maps (constant per thread)
  const int ar  = tid >> 1;           // A row 0..127
  const int akb = (tid & 1) * 16;     // A k-offset {0,16}
  const int br_t = tid >> 2;          // B(NT) row 0..63
  const int bkb  = (tid & 3) * 8;     // B(NT) k-offset {0,8,16,24}
  const int bkr  = tid >> 3;          // B(NN) k row 0..31
  const int bnb  = (tid & 7) * 8;     // B(NN) n-offset {0..56}

  const float* aSrc = A + (size_t)(m0 + ar) * lda + akb;

  auto stage = [&](int k0, int p) {
    {
      const float* src = aSrc + k0;
      float4 v0 = *(const float4*)(src + 0);
      float4 v1 = *(const float4*)(src + 4);
      float4 v2 = *(const float4*)(src + 8);
      float4 v3 = *(const float4*)(src + 12);
      pack8_store(v0, v1, &sA[p][ar][akb]);
      pack8_store(v2, v3, &sA[p][ar][akb + 8]);
      if (k0 + BK < K) __builtin_prefetch(src + BK, 0, 1); // global_prefetch_b8
    }
    if constexpr (BT) {
      const float* src = B + (size_t)(n0 + br_t) * ldb + k0 + bkb;
      float4 v0 = *(const float4*)(src + 0);
      float4 v1 = *(const float4*)(src + 4);
      pack8_store(v0, v1, &sB[p][br_t][bkb]);
    } else {
      const float* src = B + (size_t)(k0 + bkr) * ldb + n0 + bnb;
      float4 v0 = *(const float4*)(src + 0);
      float4 v1 = *(const float4*)(src + 4);
      sB[p][bnb + 0][bkr] = (bf16)v0.x; sB[p][bnb + 1][bkr] = (bf16)v0.y;
      sB[p][bnb + 2][bkr] = (bf16)v0.z; sB[p][bnb + 3][bkr] = (bf16)v0.w;
      sB[p][bnb + 4][bkr] = (bf16)v1.x; sB[p][bnb + 5][bkr] = (bf16)v1.y;
      sB[p][bnb + 6][bkr] = (bf16)v1.z; sB[p][bnb + 7][bkr] = (bf16)v1.w;
    }
  };

  v8f acc00 = {}, acc01 = {}, acc10 = {}, acc11 = {};

  stage(0, 0);
  __syncthreads();
  const int nk = K / BK;
  for (int i = 0; i < nk; ++i) {
    const int cur = i & 1;
    if (i + 1 < nk) stage((i + 1) * BK, cur ^ 1);

    v16bf a0, a1, b0, b1;
    #pragma unroll
    for (int j = 0; j < 16; ++j) {
      // A 16x32: lanes0-15 K=0..15, lanes16-31 K=16..31
      int ka = j + kg * 16;
      a0[j] = sA[cur][wm + lr][ka];
      a1[j] = sA[cur][wm + 16 + lr][ka];
      // B 32x16: lanes0-15 K=0..15, lanes16-31 K=16..31
      int kb2 = j + kg * 16;
      b0[j] = sB[cur][wn + lr][kb2];
      b1[j] = sB[cur][wn + 16 + lr][kb2];
    }
    acc00 = __builtin_amdgcn_wmma_f32_16x16x32_bf16(false, a0, false, b0, (short)0, acc00, false, false);
    acc01 = __builtin_amdgcn_wmma_f32_16x16x32_bf16(false, a0, false, b1, (short)0, acc01, false, false);
    acc10 = __builtin_amdgcn_wmma_f32_16x16x32_bf16(false, a1, false, b0, (short)0, acc10, false, false);
    acc11 = __builtin_amdgcn_wmma_f32_16x16x32_bf16(false, a1, false, b1, (short)0, acc11, false, false);
    __syncthreads();
  }

  // ---- epilogue: C/D frag: vgpr r -> row r+8*kg, col = lane&15 ----
  const int col = n0 + wn + lr;
  #pragma unroll
  for (int r = 0; r < 8; ++r) {
    int row0 = m0 + wm + kg * 8 + r;
    int row1 = row0 + 16;
    C[(size_t)row0 * ldc + col]      = acc00[r];
    C[(size_t)row0 * ldc + col + 16] = acc01[r];
    C[(size_t)row1 * ldc + col]      = acc10[r];
    C[(size_t)row1 * ldc + col + 16] = acc11[r];
  }
}

// zero-pad x_proj_w [96,2048] -> [128,2048]
__global__ __launch_bounds__(256)
void k_pad_xpw(const float* __restrict__ src, float* __restrict__ dst) {
  int idx = blockIdx.x * 256 + threadIdx.x;
  if (idx >= XDBL_P * DINNER) return;
  int r = idx / DINNER;
  int c = idx % DINNER;
  dst[idx] = (r < XDBL_W) ? src[(size_t)r * DINNER + c] : 0.0f;
}

// =====================================================================
// Depthwise causal conv (DC=4) + SiLU.  xm = xz[..., :DINNER]
// =====================================================================
__global__ __launch_bounds__(256)
void k_conv_silu(const float* __restrict__ xz, const float* __restrict__ cw,
                 const float* __restrict__ cb, float* __restrict__ u) {
  int idx = blockIdx.x * 256 + threadIdx.x;
  if (idx >= BATCH * SEQ_L * DINNER) return;
  int d  = idx % DINNER;
  int bl = idx / DINNER;
  int l  = bl % SEQ_L;
  int b  = bl / SEQ_L;
  float acc = cb[d];
  #pragma unroll
  for (int j = 0; j < DCONV; ++j) {
    int ls = l - (DCONV - 1) + j;
    if (ls >= 0)
      acc += xz[(size_t)(b * SEQ_L + ls) * (2 * DINNER) + d] * cw[d * DCONV + j];
  }
  u[idx] = siluf(acc);
}

// softplus(dt + bias) in place
__global__ __launch_bounds__(256)
void k_dt_softplus(float* __restrict__ dt, const float* __restrict__ bias) {
  int idx = blockIdx.x * 256 + threadIdx.x;
  if (idx >= BATCH * SEQ_L * DINNER) return;
  float v = dt[idx] + bias[idx % DINNER];
  dt[idx] = (v > 20.0f) ? v : log1pf(__expf(v));
}

// =====================================================================
// Selective scan: one thread per (b,d) channel; DS=16 states in regs.
// B,C read from padded x_dbl rows (stride 128, offsets 64 / 80).
// =====================================================================
__global__ __launch_bounds__(256)
void k_scan(const float* __restrict__ dt, const float* __restrict__ u,
            const float* __restrict__ xdbl, const float* __restrict__ A_log,
            float* __restrict__ ys) {
  int gid = blockIdx.x * 256 + threadIdx.x;
  if (gid >= BATCH * DINNER) return;
  int b = gid / DINNER;
  int d = gid % DINNER;
  float Ar[DSTATE], h[DSTATE];
  #pragma unroll
  for (int s = 0; s < DSTATE; ++s) { Ar[s] = -__expf(A_log[d * DSTATE + s]); h[s] = 0.0f; }
  for (int t = 0; t < SEQ_L; ++t) {
    size_t base = (size_t)(b * SEQ_L + t);
    float dtv = dt[base * DINNER + d];
    float uv  = u[base * DINNER + d];
    const float* xd = xdbl + base * XDBL_P;
    float du = dtv * uv;
    float y = 0.0f;
    #pragma unroll
    for (int s = 0; s < DSTATE; ++s) {
      h[s] = h[s] * __expf(dtv * Ar[s]) + du * xd[DTRANK + s];
      y += h[s] * xd[DTRANK + DSTATE + s];
    }
    ys[base * DINNER + d] = y;
  }
}

// y = (ys + u*D_param) * silu(z), in place on ys
__global__ __launch_bounds__(256)
void k_gate(float* __restrict__ ys, const float* __restrict__ u,
            const float* __restrict__ xz, const float* __restrict__ Dp) {
  int idx = blockIdx.x * 256 + threadIdx.x;
  if (idx >= BATCH * SEQ_L * DINNER) return;
  int d  = idx % DINNER;
  int bl = idx / DINNER;
  float z = xz[(size_t)bl * (2 * DINNER) + DINNER + d];
  ys[idx] = (ys[idx] + u[idx] * Dp[d]) * siluf(z);
}

__global__ __launch_bounds__(256)
void k_add(const float* __restrict__ a, const float* __restrict__ b,
           float* __restrict__ o, int n) {
  int idx = blockIdx.x * 256 + threadIdx.x;
  if (idx < n) o[idx] = a[idx] + b[idx];
}

// decay causal mask on scores [B, L(t), L(s)]
__global__ __launch_bounds__(256)
void k_mask(float* __restrict__ sc) {
  int idx = blockIdx.x * 256 + threadIdx.x;
  if (idx >= BATCH * SEQ_L * SEQ_L) return;
  int s = idx % SEQ_L;
  int t = (idx / SEQ_L) % SEQ_L;
  if (s < t) sc[idx] *= __expf((float)(t - 1 - s) * LN_DECAY);
  else       sc[idx] = 0.0f;
}

// out = x2 + ETA * yh
__global__ __launch_bounds__(256)
void k_final(const float* __restrict__ x2, const float* __restrict__ yh,
             float* __restrict__ out, int n) {
  int idx = blockIdx.x * 256 + threadIdx.x;
  if (idx < n) out[idx] = x2[idx] + ETA_C * yh[idx];
}

// =====================================================================
// Host-side orchestration
// =====================================================================
extern "C" void kernel_launch(void* const* d_in, const int* in_sizes, int n_in,
                              void* d_out, int out_size, void* d_ws, size_t ws_size,
                              hipStream_t stream) {
  (void)in_sizes; (void)n_in; (void)out_size; (void)ws_size;
  const float* x         = (const float*)d_in[0];
  const float* norm_w    = (const float*)d_in[1];
  const float* in_proj_w = (const float*)d_in[2];
  const float* conv_w    = (const float*)d_in[3];
  const float* conv_b    = (const float*)d_in[4];
  const float* x_proj_w  = (const float*)d_in[5];
  const float* dt_proj_w = (const float*)d_in[6];
  const float* dt_proj_b = (const float*)d_in[7];
  const float* A_log     = (const float*)d_in[8];
  const float* D_param   = (const float*)d_in[9];
  const float* out_proj_w= (const float*)d_in[10];
  const float* norm_h_w  = (const float*)d_in[11];
  const float* Wk        = (const float*)d_in[12];
  const float* Wv        = (const float*)d_in[13];
  float* out = (float*)d_out;

  const size_t BL = (size_t)BATCH * SEQ_L;   // 4096 rows
  float* ws   = (float*)d_ws;
  float* hn   = ws;                          // BL*D          (4M)
  float* xz   = hn   + BL * D_MODEL;         // BL*2*DINNER   (16M)
  float* u    = xz   + BL * 2 * DINNER;      // BL*DINNER     (8M)
  float* xdbl = u    + BL * DINNER;          // BL*128 (padded)
  float* dtb  = xdbl + BL * XDBL_P;          // BL*DINNER     (8M)
  float* ys   = dtb  + BL * DINNER;          // BL*DINNER     (8M)
  float* x2   = ys   + BL * DINNER;          // BL*D          (4M)
  float* sc   = x2   + BL * D_MODEL;         // B*L*L         (8M)
  float* xpw  = sc   + (size_t)BATCH * SEQ_L * SEQ_L; // 128*2048 padded x_proj_w
  // dead-buffer reuse:
  float* mo   = hn;                          // mamba out (hn dead after in_proj)
  float* xn   = dtb;                         // (dt dead after scan)
  float* kbuf = u;                           // (u dead after gate)
  float* vbuf = u + BL * D_MODEL;
  float* yh   = xz;                          // (xz dead after gate)

  dim3 blk(256);
  const int nIn = (int)(BL * DINNER);        // 8.4M
  const int nD  = (int)(BL * D_MODEL);       // 4.2M
  const int nSc = BATCH * SEQ_L * SEQ_L;     // 8.4M
  auto cdiv = [](int a, int b) { return (a + b - 1) / b; };

  // 0) zero-pad x_proj_w -> [128, 2048]
  k_pad_xpw<<<cdiv(XDBL_P * DINNER, 256), blk, 0, stream>>>(x_proj_w, xpw);
  // 1) h = rmsnorm(x, norm_w)
  k_rmsnorm<<<(int)BL, blk, 0, stream>>>(x, norm_w, hn, (int)BL, D_MODEL);
  // 2) xz = h @ in_proj_w^T        M=4096 N=4096 K=1024 (NT)
  k_gemm<true><<<dim3(2 * DINNER / 64, (int)BL / 128), blk, 0, stream>>>(
      hn, in_proj_w, xz, (int)BL, 2 * DINNER, D_MODEL, D_MODEL, D_MODEL, 2 * DINNER);
  // 3) u = silu(causal depthwise conv(xm) + b)
  k_conv_silu<<<cdiv(nIn, 256), blk, 0, stream>>>(xz, conv_w, conv_b, u);
  // 4) x_dbl = u @ xpw^T           M=4096 N=128 K=2048 (NT, padded)
  k_gemm<true><<<dim3(XDBL_P / 64, (int)BL / 128), blk, 0, stream>>>(
      u, xpw, xdbl, (int)BL, XDBL_P, DINNER, DINNER, DINNER, XDBL_P);
  // 5) dt = x_dbl[:, :64] @ dt_proj_w^T  (lda=128 view) M=4096 N=2048 K=64
  k_gemm<true><<<dim3(DINNER / 64, (int)BL / 128), blk, 0, stream>>>(
      xdbl, dt_proj_w, dtb, (int)BL, DINNER, DTRANK, XDBL_P, DTRANK, DINNER);
  // 6) dt = softplus(dt + bias)
  k_dt_softplus<<<cdiv(nIn, 256), blk, 0, stream>>>(dtb, dt_proj_b);
  // 7) selective scan -> ys
  k_scan<<<cdiv(BATCH * DINNER, 256), blk, 0, stream>>>(dtb, u, xdbl, A_log, ys);
  // 8) ys = (ys + u*D) * silu(z)
  k_gate<<<cdiv(nIn, 256), blk, 0, stream>>>(ys, u, xz, D_param);
  // 9) mamba_out = ys @ out_proj_w^T   M=4096 N=1024 K=2048 (NT)
  k_gemm<true><<<dim3(D_MODEL / 64, (int)BL / 128), blk, 0, stream>>>(
      ys, out_proj_w, mo, (int)BL, D_MODEL, DINNER, DINNER, DINNER, D_MODEL);
  // 10) x2 = x + mamba_out
  k_add<<<cdiv(nD, 256), blk, 0, stream>>>(x, mo, x2, nD);
  // 11) xn = rmsnorm(x2, norm_hebb_w)
  k_rmsnorm<<<(int)BL, blk, 0, stream>>>(x2, norm_h_w, xn, (int)BL, D_MODEL);
  // 12/13) k = xn @ Wk ; v = xn @ Wv   (NN) M=4096 N=1024 K=1024
  k_gemm<false><<<dim3(D_MODEL / 64, (int)BL / 128), blk, 0, stream>>>(
      xn, Wk, kbuf, (int)BL, D_MODEL, D_MODEL, D_MODEL, D_MODEL, D_MODEL);
  k_gemm<false><<<dim3(D_MODEL / 64, (int)BL / 128), blk, 0, stream>>>(
      xn, Wv, vbuf, (int)BL, D_MODEL, D_MODEL, D_MODEL, D_MODEL, D_MODEL);
  // 14) per-batch scores = xn_b @ k_b^T   M=2048 N=2048 K=1024 (NT)
  for (int b = 0; b < BATCH; ++b) {
    k_gemm<true><<<dim3(SEQ_L / 64, SEQ_L / 128), blk, 0, stream>>>(
        xn + (size_t)b * SEQ_L * D_MODEL, kbuf + (size_t)b * SEQ_L * D_MODEL,
        sc + (size_t)b * SEQ_L * SEQ_L, SEQ_L, SEQ_L, D_MODEL,
        D_MODEL, D_MODEL, SEQ_L);
  }
  // 15) apply decay causal mask
  k_mask<<<cdiv(nSc, 256), blk, 0, stream>>>(sc);
  // 16) per-batch yh = scores @ v      M=2048 N=1024 K=2048 (NN)
  for (int b = 0; b < BATCH; ++b) {
    k_gemm<false><<<dim3(D_MODEL / 64, SEQ_L / 128), blk, 0, stream>>>(
        sc + (size_t)b * SEQ_L * SEQ_L, vbuf + (size_t)b * SEQ_L * D_MODEL,
        yh + (size_t)b * SEQ_L * D_MODEL, SEQ_L, D_MODEL, SEQ_L,
        SEQ_L, D_MODEL, D_MODEL);
  }
  // 17) out = x2 + ETA * yh
  k_final<<<cdiv(nD, 256), blk, 0, stream>>>(x2, yh, out, nD);
}
